// GCNEncoder_38611755991862
// MI455X (gfx1250) — matrix operand order repaired and checked by reading, not compiled, not run on hardware
//
#include <hip/hip_runtime.h>
#include <hip/hip_bf16.h>

typedef _Float16 v16h __attribute__((ext_vector_type(16)));
typedef _Float16 v8h  __attribute__((ext_vector_type(8)));
typedef float    v8f  __attribute__((ext_vector_type(8)));

#define NNODES 20000
#define INDIM  128
#define HID    256
#define NLAYER 6

union AFrag { v16h v; v8h h[2]; };

__device__ __forceinline__ float gelu_tanh(float x) {
    float x3 = x * x * x;
    return 0.5f * x * (1.0f + tanhf(0.7978845608028654f * (x + 0.044715f * x3)));
}

__device__ __forceinline__ float wave_sum32(float v) {
    #pragma unroll
    for (int m = 16; m >= 1; m >>= 1) v += __shfl_xor(v, m, 32);
    return v;
}

// ---------------- small utility kernels ----------------

__global__ void __launch_bounds__(256) k_cvt_f16(const float* __restrict__ in,
                                                 _Float16* __restrict__ out, int n) {
    int i = blockIdx.x * 256 + threadIdx.x;
    if (i < n) out[i] = (_Float16)in[i];
}

// W: [K, Ncol] row-major f32  ->  Bt: [Ncol, K] row-major f16 (i.e. column-major W)
__global__ void __launch_bounds__(256) k_wcvt(const float* __restrict__ W,
                                              _Float16* __restrict__ Bt, int K, int Ncol) {
    int i = blockIdx.x * 256 + threadIdx.x;
    if (i >= K * Ncol) return;
    int k = i / Ncol;
    int n = i - k * Ncol;
    Bt[n * K + k] = (_Float16)W[i];
}

__global__ void __launch_bounds__(256) k_deg_init(float* __restrict__ deg, int n) {
    int i = blockIdx.x * 256 + threadIdx.x;
    if (i < n) deg[i] = 1.0f;   // self loop
}

__global__ void __launch_bounds__(256) k_deg_accum(const int* __restrict__ dst,
                                                   float* __restrict__ deg, int E) {
    int i = blockIdx.x * 256 + threadIdx.x;
    if (i < E) atomicAdd(&deg[dst[i]], 1.0f);
}

__global__ void __launch_bounds__(256) k_dinv(const float* __restrict__ deg,
                                              float* __restrict__ dinv, int n) {
    int i = blockIdx.x * 256 + threadIdx.x;
    if (i < n) dinv[i] = rsqrtf(deg[i]);
}

// ---------------- WMMA GEMM ----------------
// Z[M,256] = Ah[M,K] (f16) @ B (f16, supplied transposed as Bt[256,K]).
// If Agg != nullptr also writes Agg = Z * dinv[row]^2  (self-loop init).
// Block: 256 threads = 8 waves; block tile 64 rows x 256 cols.
// Each wave: 2 row tiles x 4 col tiles = 8 WMMAs / K-step; every B fragment
// is reused by 2 accumulators, every A fragment by 4 -> 1.5 b128 loads/WMMA.
// Tail block clamps its row base and redoes rows (identical values, no atomics),
// keeping EXEC all-ones as WMMA requires.
__global__ void __launch_bounds__(256, 2)
k_gemm_wmma(const _Float16* __restrict__ Ah, const _Float16* __restrict__ Bt,
            float* __restrict__ Z, float* __restrict__ Agg,
            const float* __restrict__ dinv, int K, int Mrows)
{
    const int lane = threadIdx.x & 31;
    const int w    = threadIdx.x >> 5;
    const int l15  = lane & 15;
    const int lhi  = lane >> 4;                        // 0 or 1

    int mblk = blockIdx.x * 64;
    if (mblk + 64 > Mrows) mblk = Mrows - 64;          // overlap-redo tail

    const int mrow  = mblk + (w >> 2) * 32;            // wave's 32-row strip
    const int nbase = (w & 3) * 64;                    // wave's 64-col strip

    v8f acc[2][4] = {};

    // A fragment: row-major [M,K]; 16-bit A 16x32 lane layout:
    // lanes 0-15 hold K = K0+[0..7] and K0+[16..23]; lanes 16-31 hold +8.
    const _Float16* aptr0 = Ah + (size_t)(mrow + l15) * K + lhi * 8;
    const _Float16* aptr1 = aptr0 + (size_t)16 * K;

    for (int k0 = 0; k0 < K; k0 += 32) {
        AFrag a0, a1;
        a0.h[0] = *(const v8h*)(aptr0 + k0);
        a0.h[1] = *(const v8h*)(aptr0 + k0 + 16);
        a1.h[0] = *(const v8h*)(aptr1 + k0);
        a1.h[1] = *(const v8h*)(aptr1 + k0 + 16);
        #pragma unroll
        for (int t = 0; t < 4; ++t) {
            // B fragment: column (nbase+t*16+l15), same K order as A
            // (lanes 0-15: K0+[0..7],[16..23]; lanes 16-31: +8).
            const _Float16* bp = Bt + (size_t)(nbase + t * 16 + l15) * K + k0 + lhi * 8;
            AFrag bf;
            bf.h[0] = *(const v8h*)bp;
            bf.h[1] = *(const v8h*)(bp + 16);
            v16h b = bf.v;
            acc[0][t] = __builtin_amdgcn_wmma_f32_16x16x32_f16(
                false, a0.v, false, b, (short)0, acc[0][t], false, false);
            acc[1][t] = __builtin_amdgcn_wmma_f32_16x16x32_f16(
                false, a1.v, false, b, (short)0, acc[1][t], false, false);
        }
    }

    // D layout: lane -> col = l15 (within tile); VGPR r -> row = r + lhi*8.
    #pragma unroll
    for (int rt = 0; rt < 2; ++rt) {
        #pragma unroll
        for (int t = 0; t < 4; ++t) {
            const int col = nbase + t * 16 + l15;
            #pragma unroll
            for (int r = 0; r < 8; ++r) {
                const int row = mrow + rt * 16 + r + lhi * 8;
                const size_t idx = (size_t)row * HID + col;
                float v = acc[rt][t][r];
                Z[idx] = v;
                if (Agg) {
                    float dn = dinv[row];
                    Agg[idx] = v * (dn * dn);
                }
            }
        }
    }
}

// ---------------- edge scatter: agg[dst] += z[src] * dinv[src]*dinv[dst] ----
__global__ void __launch_bounds__(256)
k_edge_scatter(const int* __restrict__ src, const int* __restrict__ dst,
               const float* __restrict__ dinv, const float* __restrict__ z,
               float* __restrict__ agg, int E)
{
    int e    = (blockIdx.x * 256 + threadIdx.x) >> 5;
    int lane = threadIdx.x & 31;
    if (e >= E) return;
    int s = src[e], d = dst[e];
    float wgt = dinv[s] * dinv[d];
    const float4* zp = (const float4*)(z + (size_t)s * HID);
    float* ap = agg + (size_t)d * HID;
    #pragma unroll
    for (int i = 0; i < 2; ++i) {
        float4 v = zp[lane + i * 32];
        int base = (lane + i * 32) * 4;
        atomicAdd(ap + base + 0, v.x * wgt);
        atomicAdd(ap + base + 1, v.y * wgt);
        atomicAdd(ap + base + 2, v.z * wgt);
        atomicAdd(ap + base + 3, v.w * wgt);
    }
}

// ---------------- epilogue: bias + GELU + LayerNorm (+ blend) ----------------
// mode 0 (input block):  out = LN(gelu(Y+b));  h0 = out; xcur = out; xh = f16(out)
// mode 1 (layer):        out = LN(gelu(Y+b));  xcur += 0.9*out + 0.1*h0; xh = f16(xcur)
__global__ void __launch_bounds__(256)
k_post_ln(const float* __restrict__ Yin, const float* __restrict__ bias,
          const float* __restrict__ gamma, const float* __restrict__ beta,
          float* __restrict__ h0, float* __restrict__ xcur,
          _Float16* __restrict__ xh, int mode)
{
    int lane = threadIdx.x & 31;
    int node = blockIdx.x * 8 + (threadIdx.x >> 5);
    size_t base = (size_t)node * HID;

    float t[8];
    float s1 = 0.0f, s2 = 0.0f;
    #pragma unroll
    for (int i = 0; i < 8; ++i) {
        int f = lane + i * 32;
        float v = gelu_tanh(Yin[base + f] + bias[f]);
        t[i] = v;
        s1 += v;
        s2 += v * v;
    }
    s1 = wave_sum32(s1);
    s2 = wave_sum32(s2);
    float mu   = s1 * (1.0f / (float)HID);
    float var  = s2 * (1.0f / (float)HID) - mu * mu;
    float rstd = rsqrtf(var + 1e-5f);

    #pragma unroll
    for (int i = 0; i < 8; ++i) {
        int f = lane + i * 32;
        float y = (t[i] - mu) * rstd * gamma[f] + beta[f];
        float o;
        if (mode == 0) {
            h0[base + f] = y;
            o = y;
        } else {
            o = xcur[base + f] + 0.9f * y + 0.1f * h0[base + f];
        }
        xcur[base + f] = o;
        xh[base + f]   = (_Float16)o;
    }
}

// ---------------- host-side launch ----------------

static inline char* ws_take(char*& p, size_t bytes) {
    char* r = p;
    p += (bytes + 255) & ~(size_t)255;
    return r;
}

extern "C" void kernel_launch(void* const* d_in, const int* in_sizes, int n_in,
                              void* d_out, int out_size, void* d_ws, size_t ws_size,
                              hipStream_t stream) {
    (void)n_in; (void)out_size; (void)ws_size;

    const float* x      = (const float*)d_in[0];
    const int*   ei     = (const int*)d_in[1];
    const float* W_in   = (const float*)d_in[2];
    const float* b_in   = (const float*)d_in[3];
    const float* g_in   = (const float*)d_in[4];
    const float* be_in  = (const float*)d_in[5];
    const float* Wc     = (const float*)d_in[6];
    const float* bc     = (const float*)d_in[7];
    const float* gc     = (const float*)d_in[8];
    const float* bec    = (const float*)d_in[9];

    const int E = in_sizes[1] / 2;
    const int* src = ei;
    const int* dst = ei + E;

    float* xcur = (float*)d_out;                 // fp32 running state lives in d_out

    char* p = (char*)d_ws;
    float*     deg   = (float*)    ws_take(p, (size_t)NNODES * 4);
    float*     dinv  = (float*)    ws_take(p, (size_t)NNODES * 4);
    _Float16*  xh    = (_Float16*) ws_take(p, (size_t)NNODES * HID * 2);
    _Float16*  Wh_in = (_Float16*) ws_take(p, (size_t)HID * INDIM * 2);
    _Float16*  Whc   = (_Float16*) ws_take(p, (size_t)NLAYER * HID * HID * 2);
    float*     z     = (float*)    ws_take(p, (size_t)NNODES * HID * 4);
    float*     agg   = (float*)    ws_take(p, (size_t)NNODES * HID * 4);
    float*     h0    = (float*)    ws_take(p, (size_t)NNODES * HID * 4);

    // 1) input x -> f16 [N, INDIM]
    {
        int n = NNODES * INDIM;
        k_cvt_f16<<<(n + 255) / 256, 256, 0, stream>>>(x, xh, n);
    }
    // 2) weights -> transposed f16
    k_wcvt<<<(INDIM * HID + 255) / 256, 256, 0, stream>>>(W_in, Wh_in, INDIM, HID);
    for (int l = 0; l < NLAYER; ++l) {
        k_wcvt<<<(HID * HID + 255) / 256, 256, 0, stream>>>(
            Wc + (size_t)l * HID * HID, Whc + (size_t)l * HID * HID, HID, HID);
    }
    // 3) degrees / dinv
    k_deg_init<<<(NNODES + 255) / 256, 256, 0, stream>>>(deg, NNODES);
    k_deg_accum<<<(E + 255) / 256, 256, 0, stream>>>(dst, deg, E);
    k_dinv<<<(NNODES + 255) / 256, 256, 0, stream>>>(deg, dinv, NNODES);

    const int gemm_blocks = (NNODES + 63) / 64;

    // 4) input block: z = x @ W_in ; then bias+GELU+LN -> h0, xcur, xh
    k_gemm_wmma<<<gemm_blocks, 256, 0, stream>>>(xh, Wh_in, z, nullptr, dinv, INDIM, NNODES);
    k_post_ln<<<NNODES / 8, 256, 0, stream>>>(z, b_in, g_in, be_in, h0, xcur, xh, 0);

    // 5) layers
    const int escat_blocks = (E * 32 + 255) / 256;
    for (int l = 0; l < NLAYER; ++l) {
        k_gemm_wmma<<<gemm_blocks, 256, 0, stream>>>(
            xh, Whc + (size_t)l * HID * HID, z, agg, dinv, HID, NNODES);
        k_edge_scatter<<<escat_blocks, 256, 0, stream>>>(src, dst, dinv, z, agg, E);
        k_post_ln<<<NNODES / 8, 256, 0, stream>>>(
            agg, bc + (size_t)l * HID, gc + (size_t)l * HID, bec + (size_t)l * HID,
            h0, xcur, xh, 1);
    }
}
